// SpatialRootLayer_49916109914255
// MI455X (gfx1250) — compile-verified, matrix operand order checked by reading, not run on hardware
//
#include <hip/hip_runtime.h>
#include <cmath>

typedef __attribute__((ext_vector_type(16))) _Float16 v16h;
typedef __attribute__((ext_vector_type(8)))  float    v8f;

namespace {
constexpr int BATCH = 128;
constexpr int NOUT  = 128;
constexpr int KDIM  = 16384;            // 16*32*32
constexpr int KBLKS = KDIM / 32;        // 512 K-blocks of 32
constexpr int KCHUNKS = 16;             // grid split along K
constexpr int KB_PER_CHUNK = KBLKS / KCHUNKS; // 32
}

// CDNA5 async global->LDS copy, 16 bytes per lane (tracked by ASYNCcnt).
__device__ __forceinline__ void async_ld_b128(_Float16* lds, const char* gsrc) {
    auto l3 = (__attribute__((address_space(3))) _Float16*)lds;
    unsigned lds_off = (unsigned)(unsigned long long)l3;
    asm volatile("global_load_async_to_lds_b128 %0, %1, off"
                 :: "v"(lds_off), "v"(gsrc)
                 : "memory");
}

__global__ void __launch_bounds__(256)
zero_f32(float* __restrict__ p, int n) {
    int i = blockIdx.x * 256 + threadIdx.x;
    if (i < n) p[i] = 0.0f;
}

// One block per row. Pass 1: row max. Pass 2: write exp(v - rowmax) as f16
// into the WMMA 16-bit A-fragment packed layout:
//   pack[(tile*KBLKS + kblk)*32 + lane][h],  lane = (row&15) | (((k>>3)&1)<<4),
//   h = (k&7) | (((k>>4)&1)<<3)   (inverse of ISA k = (h&7)+16*(h>>3)+8*(lane>>4))
// Optionally emits rowmax and log(sum exp).
__global__ void __launch_bounds__(256)
row_exp_pack(const float* __restrict__ src, _Float16* __restrict__ pack,
             float* __restrict__ rowmax_out, float* __restrict__ logz_out) {
    __shared__ float red[256];
    const int row = blockIdx.x;
    const int tid = threadIdx.x;
    const float* srow = src + (size_t)row * KDIM;

    float m = -INFINITY;
    for (int k = tid; k < KDIM; k += 256) m = fmaxf(m, srow[k]);
    red[tid] = m;
    __syncthreads();
    for (int s = 128; s > 0; s >>= 1) {
        if (tid < s) red[tid] = fmaxf(red[tid], red[tid + s]);
        __syncthreads();
    }
    const float rmax = red[0];
    __syncthreads();

    const int tile = row >> 4;
    const int mlow = row & 15;
    float lsum = 0.0f;
    for (int k = tid; k < KDIM; k += 256) {
        float e = __expf(srow[k] - rmax);
        lsum += e;
        int kb   = k >> 5;
        int lane = mlow | (((k >> 3) & 1) << 4);
        int h    = (k & 7) | (((k >> 4) & 1) << 3);
        pack[(size_t)(((tile * KBLKS + kb) << 5) + lane) * 16 + h] = (_Float16)e;
    }
    if (logz_out) {
        red[tid] = lsum;
        __syncthreads();
        for (int s = 128; s > 0; s >>= 1) {
            if (tid < s) red[tid] += red[tid + s];
            __syncthreads();
        }
        if (tid == 0) logz_out[row] = __logf(red[0]);
    }
    if (rowmax_out && tid == 0) rowmax_out[row] = rmax;
}

// 8 waves / workgroup; wave w owns M-band [16w,16w+16) x all 128 columns.
// grid.x = KCHUNKS; each block accumulates its K slice and atomically merges.
// B[kb] (8 KB, fragment-packed) is triple-buffered in LDS via async loads at
// prefetch distance 2; A fragments are register double-buffered; all 8 B
// fragments are loaded from LDS in one batch before the 8 WMMAs so the ds
// clause overlaps the XDL train. One barrier per kb: the copy for kb+2
// targets buf[(i-1)%3], whose readers all passed this iteration's barrier.
__global__ void __launch_bounds__(256)
gemm_wmma_f16(const _Float16* __restrict__ Ap, const _Float16* __restrict__ Bp,
              float* __restrict__ S) {
    __shared__ __align__(32) _Float16 lbuf[3][8 * 512];   // 3 x 8 KB B staging

    const int tid  = threadIdx.x;
    const int lane = tid & 31;
    const int wv   = tid >> 5;                 // 0..7 -> M tile
    const int kb0  = blockIdx.x * KB_PER_CHUNK;

    // copy mapping: thread t moves tile n = t>>5, lane l = t&31 (32 B)
    const int cn = tid >> 5;
    const int cl = tid & 31;

    v8f acc[8] = {};

    {   // prologue: stage B[kb0] -> buf0, B[kb0+1] -> buf1
        const char* s0 =
            (const char*)(Bp + (size_t)(((cn * KBLKS + kb0) << 5) + cl) * 16);
        async_ld_b128(&lbuf[0][tid * 16], s0);
        async_ld_b128(&lbuf[0][tid * 16 + 8], s0 + 16);
        const char* s1 =
            (const char*)(Bp + (size_t)(((cn * KBLKS + kb0 + 1) << 5) + cl) * 16);
        async_ld_b128(&lbuf[1][tid * 16], s1);
        async_ld_b128(&lbuf[1][tid * 16 + 8], s1 + 16);
    }
    v16h a_cur = *(const v16h*)(Ap +
                 (size_t)((((wv * KBLKS + kb0) << 5) + lane) * 16));

    for (int i = 0; i < KB_PER_CHUNK; ++i) {
        const int kb = kb0 + i;

        // current buffer's pair was issued 2 iterations back; exactly one
        // newer pair is outstanding (none on the last iteration).
        if (i < KB_PER_CHUNK - 1) {
            asm volatile("s_wait_asynccnt 0x2" ::: "memory");
        } else {
            asm volatile("s_wait_asynccnt 0x0" ::: "memory");
        }
        __syncthreads();   // buf[i%3] visible; buf[(i+2)%3] readers are done

        if (i + 2 < KB_PER_CHUNK) {
            const char* src = (const char*)
                (Bp + (size_t)(((cn * KBLKS + kb + 2) << 5) + cl) * 16);
            _Float16* dst = &lbuf[(i + 2) % 3][tid * 16];
            async_ld_b128(dst, src);
            async_ld_b128(dst + 8, src + 16);
        }

        v16h a_nxt = a_cur;
        if (i + 1 < KB_PER_CHUNK) {
            a_nxt = *(const v16h*)(Ap +
                    (size_t)((((wv * KBLKS + kb + 1) << 5) + lane) * 16));
        }

        const _Float16* bb = &lbuf[i % 3][0];
        v16h bf[8];
#pragma unroll
        for (int n = 0; n < 8; ++n) {
            bf[n] = *(const v16h*)&bb[((n << 5) | lane) * 16];
        }
#pragma unroll
        for (int n = 0; n < 8; ++n) {
            acc[n] = __builtin_amdgcn_wmma_f32_16x16x32_f16(
                false, a_cur, false, bf[n], (short)0, acc[n], false, false);
        }
        a_cur = a_nxt;
    }

    // D layout: VGPR r, lanes 0-15 -> M=r, lanes 16-31 -> M=r+8; N = lane&15.
    const int mbase = wv * 16 + ((lane >> 4) << 3);
    const int cbase = lane & 15;
#pragma unroll
    for (int n = 0; n < 8; ++n) {
#pragma unroll
        for (int r = 0; r < 8; ++r) {
            unsafeAtomicAdd(&S[(mbase + r) * NOUT + (n * 16 + cbase)], acc[n][r]);
        }
    }
}

__global__ void __launch_bounds__(256)
finalize(const float* __restrict__ S, const float* __restrict__ mx,
         const float* __restrict__ logz, float* __restrict__ out) {
    int i = blockIdx.x * 256 + threadIdx.x;
    if (i >= BATCH * NOUT) return;
    int b = i >> 7;
    int o = i & (NOUT - 1);
    out[i] = mx[b] + __logf(S[i]) - logz[o];
}

extern "C" void kernel_launch(void* const* d_in, const int* in_sizes, int n_in,
                              void* d_out, int out_size, void* d_ws, size_t ws_size,
                              hipStream_t stream) {
    const float* x = (const float*)d_in[0];   // [128,16,32,32] f32
    const float* w = (const float*)d_in[1];   // [128,16384]   f32
    float* out = (float*)d_out;               // [128,128]     f32

    char* ws = (char*)d_ws;
    float* mx   = (float*)ws;                 // 128 f32
    float* logz = mx + BATCH;                 // 128 f32
    float* S    = logz + NOUT;                // 128*128 f32
    // 66560 bytes used above (32B aligned); f16 packed operands follow.
    _Float16* Ap = (_Float16*)(ws + 66560);           // 128*16384 f16 = 4 MB
    _Float16* Bp = Ap + (size_t)BATCH * KDIM;         // 128*16384 f16 = 4 MB

    zero_f32<<<(BATCH * NOUT + 255) / 256, 256, 0, stream>>>(S, BATCH * NOUT);
    row_exp_pack<<<BATCH, 256, 0, stream>>>(x, Ap, mx, nullptr);
    row_exp_pack<<<NOUT, 256, 0, stream>>>(w, Bp, nullptr, logz);
    gemm_wmma_f16<<<KCHUNKS, 256, 0, stream>>>(Ap, Bp, S);
    finalize<<<(BATCH * NOUT + 255) / 256, 256, 0, stream>>>(S, mx, logz, out);
}
